// EncoderLayer_44521630990781
// MI455X (gfx1250) — matrix operand order repaired and checked by reading, not compiled, not run on hardware
//
#include <hip/hip_runtime.h>
#include <cstdint>

#define B_   8
#define S_   1024
#define H_   512
#define NH_  8
#define DH_  64
#define FFN_ 2048

typedef __bf16 bf16;
typedef __attribute__((ext_vector_type(16))) __bf16 v16bf;
typedef __attribute__((ext_vector_type(8)))  float  v8f;

union FragAB { v16bf v; uint4 q[2]; bf16 h[16]; };
union FragC  { v8f v; float f[8]; };

__device__ __forceinline__ v8f wmma_bf16(v16bf a, v16bf b, v8f c) {
  return __builtin_amdgcn_wmma_f32_16x16x32_bf16(
      /*neg_a=*/false, a, /*neg_b=*/false, b,
      /*c_mod=*/(short)0, c, /*reuse_a=*/false, /*reuse_b=*/false);
}

// A-fragment (16x32 bf16, MxK), source row-major with stride `strideH` halves.
// ISA 7.12.2: lanes 0-15 row M=lane, K = {0..7, 16..23};
//             lanes 16-31 row M=lane-16, K = {8..15, 24..31}.
__device__ __forceinline__ v16bf frag_a(const bf16* p, int row0, int strideH, int koff) {
  const int lane = threadIdx.x & 31;
  const bf16* s = p + (size_t)(row0 + (lane & 15)) * strideH + koff + (lane >> 4) * 8;
  FragAB f;
  f.q[0] = *(const uint4*)(s);
  f.q[1] = *(const uint4*)(s + 16);
  return f.v;
}

// B-fragment (32x16 bf16, KxN), source stored TRANSPOSED as [col][K] with
// stride `strideH` halves. lane = column N, K = {0..7,16..23} (+8 for lane>=16).
__device__ __forceinline__ v16bf frag_bT(const bf16* p, int col0, int strideH, int koff) {
  const int lane = threadIdx.x & 31;
  const bf16* s = p + (size_t)(col0 + (lane & 15)) * strideH + koff + (lane >> 4) * 8;
  FragAB f;
  f.q[0] = *(const uint4*)(s);
  f.q[1] = *(const uint4*)(s + 16);
  return f.v;
}

// ---- CDNA5 async global->LDS staging (ASYNCcnt path) ----------------------
__device__ __forceinline__ unsigned lds_u32(const void* p) {
  return (unsigned)(uintptr_t)p;  // low 32 bits of LDS-aperture addr = LDS offset
}
__device__ __forceinline__ void async_b128(unsigned lds, const bf16* g) {
  asm volatile("global_load_async_to_lds_b128 %0, %1, off"
               :: "v"(lds), "v"((unsigned long long)(uintptr_t)g)
               : "memory");
}
__device__ __forceinline__ void wait_async0() {
  asm volatile("s_wait_asynccnt 0x0" ::: "memory");
}

// ---------------------------------------------------------------- utilities
// fp32 [K][N] -> bf16 transposed [N][K]
__global__ __launch_bounds__(256) void cvtT_kernel(const float* __restrict__ src,
                                                   bf16* __restrict__ dst,
                                                   int K, int N) {
  int idx = blockIdx.x * 256 + threadIdx.x;
  if (idx < K * N) {
    int k = idx / N, n = idx % N;
    dst[(size_t)n * K + k] = (bf16)src[idx];
  }
}

// Combined masked bias: bm = mask ? bias : -1e9, as bf16. Loads hoisted into
// registers first so the select lowers to v_cndmask (no exec-mask branch).
__global__ __launch_bounds__(256) void bm_kernel(const float* __restrict__ bias,
                                                 const int* __restrict__ mask,
                                                 bf16* __restrict__ bm, int n) {
  int i = blockIdx.x * 256 + threadIdx.x;
  if (i < n) {
    float bv = bias[i];
    int   mv = mask[i];
    float r = mv ? bv : -1e9f;
    bm[i] = (bf16)r;
  }
}

// One wave per row (wave32). H=512 -> 16 elems/lane.
__global__ __launch_bounds__(256) void ln_kernel(const float* __restrict__ x,
                                                 const float* __restrict__ g,
                                                 const float* __restrict__ bta,
                                                 bf16* __restrict__ y) {
  const int row  = blockIdx.x * 8 + (threadIdx.x >> 5);
  const int lane = threadIdx.x & 31;
  const float* xr = x + (size_t)row * H_;
  float v[16];
  float s = 0.f;
#pragma unroll
  for (int i = 0; i < 16; i++) { v[i] = xr[i * 32 + lane]; s += v[i]; }
#pragma unroll
  for (int m = 16; m >= 1; m >>= 1) s += __shfl_xor(s, m, 32);
  const float mu = s * (1.0f / H_);
  float var = 0.f;
#pragma unroll
  for (int i = 0; i < 16; i++) { float d = v[i] - mu; var += d * d; }
#pragma unroll
  for (int m = 16; m >= 1; m >>= 1) var += __shfl_xor(var, m, 32);
  const float rstd = rsqrtf(var * (1.0f / H_) + 1e-5f);
  bf16* yr = y + (size_t)row * H_;
#pragma unroll
  for (int i = 0; i < 16; i++) {
    int c = i * 32 + lane;
    yr[c] = (bf16)((v[i] - mu) * rstd * g[c] + bta[c]);
  }
}

// ---------------------------------------------------------------- WMMA GEMM
enum { EPI_QKV = 0, EPI_OPROJ = 1, EPI_FFN1 = 2, EPI_FFN2 = 3 };

// 128x128 tile per block, 8 waves, each wave 32x64 (2 A-frags x 4 B-frags =
// 8 WMMAs per K-step). A: [M,K] bf16 row-major. WT: [N,K] bf16 (transposed).
// Double-buffered global_load_async_to_lds_b128 staging. Dims multiples of 128.
template <int EPI>
__global__ __launch_bounds__(256) void gemm_kernel(
    const bf16* __restrict__ A, const bf16* __restrict__ WT,
    int M, int N, int K,
    const float* __restrict__ bias, const float* __restrict__ res,
    float* __restrict__ outF, bf16* __restrict__ outB) {
  const int z = blockIdx.z;
  if (EPI == EPI_QKV) { WT += (size_t)z * K * N; bias += z * N; }

  __shared__ __align__(16) bf16 sA[2][128 * 40];   // [row][k], stride 40 halves
  __shared__ __align__(16) bf16 sBT[2][128 * 40];  // [col][k], stride 40 halves
  const unsigned BUFB = 128 * 40 * 2;              // buffer stride in bytes

  const int tid  = threadIdx.x;
  const int wave = tid >> 5;
  const int lane = tid & 31;
  const int m0 = blockIdx.y * 128, n0 = blockIdx.x * 128;
  const int wm = (wave & 3) * 32;
  const int wn = (wave >> 2) * 64;

  FragC c[2][4] = {};

  const int sr = tid >> 1;          // 0..127 (tile row / col)
  const int scol = (tid & 1) * 16;  // 0 or 16 (k offset, halves)
  const bf16* aRow = A  + (size_t)(m0 + sr) * K + scol;
  const bf16* bRow = WT + (size_t)(n0 + sr) * K + scol;
  const unsigned ldsA0 = lds_u32(&sA[0][sr * 40 + scol]);
  const unsigned ldsB0 = lds_u32(&sBT[0][sr * 40 + scol]);

  // prologue: fill buffer 0
  async_b128(ldsA0,      aRow);
  async_b128(ldsA0 + 16, aRow + 8);
  async_b128(ldsB0,      bRow);
  async_b128(ldsB0 + 16, bRow + 8);
  wait_async0();
  __syncthreads();

  int cur = 0;
  for (int k0 = 0; k0 < K; k0 += 32) {
    const int nxt = cur ^ 1;
    if (k0 + 32 < K) {  // prefetch next K-tile into the idle buffer
      const unsigned la = ldsA0 + (unsigned)nxt * BUFB;
      const unsigned lb = ldsB0 + (unsigned)nxt * BUFB;
      async_b128(la,      aRow + k0 + 32);
      async_b128(la + 16, aRow + k0 + 40);
      async_b128(lb,      bRow + k0 + 32);
      async_b128(lb + 16, bRow + k0 + 40);
    }
    const bf16* cA = sA[cur];
    const bf16* cB = sBT[cur];
    v16bf a0 = frag_a(cA, wm, 40, 0);
    v16bf a1 = frag_a(cA, wm + 16, 40, 0);
#pragma unroll
    for (int j = 0; j < 4; j++) {
      v16bf b = frag_bT(cB, wn + j * 16, 40, 0);
      c[0][j].v = wmma_bf16(a0, b, c[0][j].v);
      c[1][j].v = wmma_bf16(a1, b, c[1][j].v);
    }
    wait_async0();    // next tile landed
    __syncthreads();  // all waves done with `cur`, new data visible
    cur = nxt;
  }

  const int ln = lane & 15, hi = lane >> 4;
#pragma unroll
  for (int mi = 0; mi < 2; mi++) {
#pragma unroll
    for (int e = 0; e < 8; e++) {
      const int gm = m0 + wm + mi * 16 + e + 8 * hi;  // C: M = vgpr + 8*(lane>=16)
#pragma unroll
      for (int j = 0; j < 4; j++) {
        const int gn = n0 + wn + j * 16 + ln;
        float acc = c[mi][j].f[e];
        if (EPI == EPI_QKV) {
          float val = acc + bias[gn];
          int hh = gn >> 6, d = gn & 63;
          int bb = gm >> 10, ss = gm & 1023;
          size_t base = (size_t)z * ((size_t)B_ * S_ * H_);
          size_t dst;
          if (z == 2)  // V stored transposed per head: [B][NH][DH][S]
            dst = base + (((size_t)bb * NH_ + hh) * DH_ + d) * S_ + ss;
          else         // Q,K: [B][NH][S][DH]
            dst = base + (((size_t)bb * NH_ + hh) * S_ + ss) * DH_ + d;
          outB[dst] = (bf16)val;
        } else if (EPI == EPI_OPROJ) {
          size_t idx = (size_t)gm * N + gn;
          outF[idx] = acc + bias[gn] + res[idx];
        } else if (EPI == EPI_FFN1) {
          float u = acc + bias[gn];
          outB[(size_t)gm * N + gn] =
              (bf16)(0.5f * u * (1.0f + erff(u * 0.70710678118654752f)));
        } else {  // EPI_FFN2
          size_t idx = (size_t)gm * N + gn;
          outF[idx] = acc + bias[gn] + res[idx];
        }
      }
    }
  }
}

// --------------------------------------------------------- flash attention
// Block = 128 threads (4 waves), one (b,h), 64 queries. Each wave owns a
// 16-query tile; online softmax over kv in steps of 32. K read directly from
// global ([key][d] == transposed-B layout); V read directly from global in
// pre-transposed [d][key] layout. bm = premasked bias (bf16), single load +
// fma_mix per score element. Row sums of exp(P) are accumulated on the WMMA
// pipe via an all-ones B fragment (same C row layout as O -> same alpha
// rescale); only the row-max reduction uses cross-lane shuffles.
__global__ __launch_bounds__(128) void attn_kernel(
    const bf16* __restrict__ q, const bf16* __restrict__ k,
    const bf16* __restrict__ vT, const bf16* __restrict__ bm,
    bf16* __restrict__ o) {
  const int b = blockIdx.z, h = blockIdx.y;
  const int wave = threadIdx.x >> 5, lane = threadIdx.x & 31;
  const int qbase = blockIdx.x * 64 + wave * 16;
  const size_t bhoff = (size_t)(b * NH_ + h) * S_ * DH_;
  const bf16* qp  = q + bhoff;
  const bf16* kp  = k + bhoff;
  const bf16* vtp = vT + bhoff;  // [DH][S]

  __shared__ __align__(16) bf16 sP[4][16 * 40];  // per-wave P tile [m][kk]

  const int ln = lane & 15, hi = lane >> 4;

  v16bf aq0 = frag_a(qp, qbase, DH_, 0);
  v16bf aq1 = frag_a(qp, qbase, DH_, 32);

  FragAB onesF;
#pragma unroll
  for (int i = 0; i < 16; i++) onesF.h[i] = (bf16)1.0f;

  FragC o0{}, o1{}, o2{}, o3{}, osum{};
  float mrow[8];
#pragma unroll
  for (int i = 0; i < 8; i++) mrow[i] = -3.0e38f;
  const float scale = 0.125f;  // DH^-0.5

  for (int kv = 0; kv < S_; kv += 32) {
    // scores for two 16-key tiles (Kt fragments direct from global)
    float f0[8], f1[8];
#pragma unroll
    for (int t = 0; t < 2; t++) {
      int kb = kv + t * 16;
      FragC sc{};
      sc.v = wmma_bf16(aq0, frag_bT(kp, kb, DH_, 0), sc.v);
      sc.v = wmma_bf16(aq1, frag_bT(kp, kb, DH_, 32), sc.v);
      float* fo = t ? f1 : f0;
      const int key = kb + ln;
#pragma unroll
      for (int i = 0; i < 8; i++) {
        int qm = qbase + i + 8 * hi;
        float bmv = (float)bm[((size_t)b * S_ + qm) * S_ + key];
        fo[i] = fmaf(sc.f[i], scale, bmv);
      }
    }
    // online softmax: row-max via shuffles; exp; rescale accumulators
#pragma unroll
    for (int i = 0; i < 8; i++) {
      float tm = fmaxf(f0[i], f1[i]);
#pragma unroll
      for (int m = 8; m >= 1; m >>= 1) tm = fmaxf(tm, __shfl_xor(tm, m, 32));
      float mn = fmaxf(mrow[i], tm);
      float alpha = __expf(mrow[i] - mn);
      mrow[i] = mn;
      float p0 = __expf(f0[i] - mn), p1 = __expf(f1[i] - mn);
      int qm16 = i + 8 * hi;
      sP[wave][qm16 * 40 + ln]      = (bf16)p0;
      sP[wave][qm16 * 40 + 16 + ln] = (bf16)p1;
      o0.f[i] *= alpha; o1.f[i] *= alpha; o2.f[i] *= alpha; o3.f[i] *= alpha;
      osum.f[i] *= alpha;
    }
    __syncthreads();  // P C-layout -> A-layout round trip through LDS
    v16bf ap = frag_a(sP[wave], 0, 40, 0);
    osum.v = wmma_bf16(ap, onesF.v, osum.v);  // row sums on the matrix pipe
    o0.v = wmma_bf16(ap, frag_bT(vtp, 0,  S_, kv), o0.v);
    o1.v = wmma_bf16(ap, frag_bT(vtp, 16, S_, kv), o1.v);
    o2.v = wmma_bf16(ap, frag_bT(vtp, 32, S_, kv), o2.v);
    o3.v = wmma_bf16(ap, frag_bT(vtp, 48, S_, kv), o3.v);
  }
  // normalize + store to o[B*S][H] at column h*64 (bf16 for o-proj GEMM)
#pragma unroll
  for (int i = 0; i < 8; i++) {
    float inv = 1.0f / osum.f[i];  // every column of osum holds the row sum
    int qm = qbase + i + 8 * hi;
    size_t row = ((size_t)b * S_ + qm) * H_ + h * DH_;
    o[row + ln]      = (bf16)(o0.f[i] * inv);
    o[row + 16 + ln] = (bf16)(o1.f[i] * inv);
    o[row + 32 + ln] = (bf16)(o2.f[i] * inv);
    o[row + 48 + ln] = (bf16)(o3.f[i] * inv);
  }
}

// ------------------------------------------------------------------- launch
extern "C" void kernel_launch(void* const* d_in, const int* in_sizes, int n_in,
                              void* d_out, int out_size, void* d_ws, size_t ws_size,
                              hipStream_t stream) {
  (void)in_sizes; (void)n_in; (void)out_size; (void)ws_size;
  const float* x     = (const float*)d_in[0];
  const float* abias = (const float*)d_in[1];
  const int*   gmask = (const int*)d_in[2];
  const float* ln1_g = (const float*)d_in[3];
  const float* ln1_b = (const float*)d_in[4];
  const float* Wq = (const float*)d_in[5];
  const float* bq = (const float*)d_in[6];
  const float* Wk = (const float*)d_in[7];
  const float* bk = (const float*)d_in[8];
  const float* Wv = (const float*)d_in[9];
  const float* bv = (const float*)d_in[10];
  const float* Wo = (const float*)d_in[11];
  const float* bo = (const float*)d_in[12];
  const float* ln2_g = (const float*)d_in[13];
  const float* ln2_b = (const float*)d_in[14];
  const float* W1 = (const float*)d_in[15];
  const float* b1 = (const float*)d_in[16];
  const float* W2 = (const float*)d_in[17];
  const float* b2 = (const float*)d_in[18];

  char* p = (char*)d_ws;
  auto alloc = [&](size_t bytes) {
    char* r = p;
    p += (bytes + 255) & ~(size_t)255;
    return r;
  };
  const size_t M = (size_t)B_ * S_;
  bf16*  wqkvT   = (bf16*)alloc((size_t)3 * H_ * H_ * 2);   // [3][N=H][K=H]
  bf16*  woT     = (bf16*)alloc((size_t)H_ * H_ * 2);       // [N=H][K=H]
  bf16*  w1T     = (bf16*)alloc((size_t)H_ * FFN_ * 2);     // [N=FFN][K=H]
  bf16*  w2T     = (bf16*)alloc((size_t)FFN_ * H_ * 2);     // [N=H][K=FFN]
  float* qkvBias = (float*)alloc((size_t)3 * H_ * 4);
  bf16*  bm      = (bf16*)alloc((size_t)B_ * S_ * S_ * 2);  // premasked bias
  bf16*  ybf     = (bf16*)alloc(M * H_ * 2);
  bf16*  qkv     = (bf16*)alloc((size_t)3 * M * H_ * 2);    // q,k:[B][NH][S][DH]; v:[B][NH][DH][S]
  bf16*  obf     = (bf16*)alloc(M * H_ * 2);
  float* x1      = (float*)alloc(M * H_ * 4);
  bf16*  y2bf    = (bf16*)alloc(M * H_ * 2);
  bf16*  hbuf    = (bf16*)alloc(M * FFN_ * 2);

  const int HH = H_ * H_, HF = H_ * FFN_;
  cvtT_kernel<<<dim3((HH + 255) / 256), 256, 0, stream>>>(Wq, wqkvT, H_, H_);
  cvtT_kernel<<<dim3((HH + 255) / 256), 256, 0, stream>>>(Wk, wqkvT + HH, H_, H_);
  cvtT_kernel<<<dim3((HH + 255) / 256), 256, 0, stream>>>(Wv, wqkvT + 2 * HH, H_, H_);
  cvtT_kernel<<<dim3((HH + 255) / 256), 256, 0, stream>>>(Wo, woT, H_, H_);
  cvtT_kernel<<<dim3((HF + 255) / 256), 256, 0, stream>>>(W1, w1T, H_, FFN_);
  cvtT_kernel<<<dim3((HF + 255) / 256), 256, 0, stream>>>(W2, w2T, FFN_, H_);
  hipMemcpyAsync(qkvBias,          bq, H_ * 4, hipMemcpyDeviceToDevice, stream);
  hipMemcpyAsync(qkvBias + H_,     bk, H_ * 4, hipMemcpyDeviceToDevice, stream);
  hipMemcpyAsync(qkvBias + 2 * H_, bv, H_ * 4, hipMemcpyDeviceToDevice, stream);

  // premasked attention bias (branchless bf16 combine)
  {
    int n = B_ * S_ * S_;
    bm_kernel<<<dim3((n + 255) / 256), 256, 0, stream>>>(abias, gmask, bm, n);
  }
  // LN1 -> bf16
  ln_kernel<<<dim3((int)(M / 8)), 256, 0, stream>>>(x, ln1_g, ln1_b, ybf);
  // fused Q/K/V projections (blockIdx.z picks matrix; V written transposed)
  gemm_kernel<EPI_QKV><<<dim3(H_ / 128, (int)(M / 128), 3), 256, 0, stream>>>(
      ybf, wqkvT, (int)M, H_, H_, qkvBias, nullptr, nullptr, qkv);
  // flash attention
  attn_kernel<<<dim3(S_ / 64, NH_, B_), 128, 0, stream>>>(
      qkv, qkv + M * H_, qkv + 2 * M * H_, bm, obf);
  // O projection + residual -> x1
  gemm_kernel<EPI_OPROJ><<<dim3(H_ / 128, (int)(M / 128), 1), 256, 0, stream>>>(
      obf, woT, (int)M, H_, H_, bo, x, x1, nullptr);
  // LN2 -> bf16
  ln_kernel<<<dim3((int)(M / 8)), 256, 0, stream>>>(x1, ln2_g, ln2_b, y2bf);
  // FFN1 + exact GELU -> bf16
  gemm_kernel<EPI_FFN1><<<dim3(FFN_ / 128, (int)(M / 128), 1), 256, 0, stream>>>(
      y2bf, w1T, (int)M, FFN_, H_, b1, nullptr, nullptr, hbuf);
  // FFN2 + residual -> out (fp32)
  gemm_kernel<EPI_FFN2><<<dim3(H_ / 128, (int)(M / 128), 1), 256, 0, stream>>>(
      hbuf, w2T, (int)M, H_, FFN_, b2, x1, (float*)d_out, nullptr);
}
